// PTBModel_7653631721847
// MI455X (gfx1250) — matrix operand order repaired and checked by reading, not compiled, not run on hardware
//
#include <hip/hip_runtime.h>
#include <hip/hip_bf16.h>

#define BATCH 32
#define SEQ   256
#define EMB   1024
#define HID   1024
#define VOCAB 10000

typedef __attribute__((ext_vector_type(16))) __bf16 v16bf;
typedef __attribute__((ext_vector_type(8)))  __bf16 v8bf;
typedef __attribute__((ext_vector_type(8)))  float  v8f;

// ---------------------------------------------------------------------------
// WMMA operand loaders (v_wmma_f32_16x16x32_bf16, wave32 layouts per ISA 7.12.2)
// A (16x32, MxK): lanes 0-15 -> M=lane, lanes 16-31 -> M=lane-16.
//   elems 0..7  = K in [hi*8,   hi*8+7]      (VGPRs 0-3)
//   elems 8..15 = K in [16+hi*8, 16+hi*8+7]  (VGPRs 4-7)
// B (32x16, KxN): N = lane&15; elems 0..7 = K hi*8+0..7, elems 8..15 = K 16+hi*8+0..7.
// C/D (16x16 f32): vgpr r, lane -> M = r + (lane>=16 ? 8 : 0), N = lane&15.
// ---------------------------------------------------------------------------
__device__ __forceinline__ v16bf load_a_tile(const __bf16* A, int lda,
                                             int row0, int k0, int lane) {
  int lo = lane & 15, hi = (lane >> 4) & 1;
  const __bf16* p = A + (size_t)(row0 + lo) * lda + (k0 + hi * 8);
  v8bf a0 = *(const v8bf*)(p);        // K = hi*8 + 0..7
  v8bf a1 = *(const v8bf*)(p + 16);   // K = 16 + hi*8 + 0..7
  v16bf r;
#pragma unroll
  for (int i = 0; i < 8; ++i) { r[i] = a0[i]; r[i + 8] = a1[i]; }
  return r;
}

__device__ __forceinline__ v16bf load_b_tile(const __bf16* Bm, int ldb,
                                             int k0, int n0, int lane) {
  int lo = lane & 15, hi = (lane >> 4) & 1;
  const __bf16* p = Bm + (size_t)(k0 + hi * 8) * ldb + n0 + lo;
  v16bf r;
#pragma unroll
  for (int i = 0; i < 8; ++i) { r[i] = p[(size_t)i * ldb]; r[i + 8] = p[(size_t)(16 + i) * ldb]; }
  return r;
}

// ---------------------------------------------------------------------------
// CDNA5 async Global->LDS copy (16 bytes), tracked by ASYNCcnt.
// LDS destination = low 32 bits of the generic pointer (addrspace(3) offset).
// ---------------------------------------------------------------------------
__device__ __forceinline__ void async_copy_b128(unsigned lds_off,
                                                const __bf16* gsrc) {
  asm volatile("global_load_async_to_lds_b128 %0, %1, off"
               :: "v"(lds_off), "v"(gsrc) : "memory");
}
__device__ __forceinline__ void wait_async0() {
  asm volatile("s_wait_asynccnt 0x0" ::: "memory");
}

// ---------------------------------------------------------------------------
// Generic bf16 GEMM + bias -> f32:  C[M,N] = A[M,K] @ B[K,N] + bias[N]
// One wave per 16x16 tile (used for the input-projection GEMM).
// ---------------------------------------------------------------------------
__global__ void k_gemm_bf16_bias(const __bf16* __restrict__ A,
                                 const __bf16* __restrict__ Bm,
                                 const float* __restrict__ bias,
                                 float* __restrict__ C,
                                 int M, int N, int K) {
  int wave = (int)((blockIdx.x * blockDim.x + threadIdx.x) >> 5);
  int lane = threadIdx.x & 31;
  int ntn = N >> 4;
  int total = (M >> 4) * ntn;
  if (wave >= total) return;                 // wave-uniform branch
  int mt = wave / ntn, nt = wave % ntn;
  int row0 = mt << 4, n0 = nt << 4;

  v8f acc = {};
  for (int k0 = 0; k0 < K; k0 += 32) {
    v16bf a = load_a_tile(A, K, row0, k0, lane);
    v16bf b = load_b_tile(Bm, N, k0, n0, lane);
    acc = __builtin_amdgcn_wmma_f32_16x16x32_bf16(false, a, false, b,
                                                  (short)0, acc, false, false);
  }
  int lo = lane & 15, hi = (lane >> 4) & 1;
  float bv = bias[n0 + lo];
#pragma unroll
  for (int r = 0; r < 8; ++r)
    C[(size_t)(row0 + r + hi * 8) * N + (n0 + lo)] = acc[r] + bv;
}

// ---------------------------------------------------------------------------
// LDS-tiled dense GEMM (dominant kernel, ~168 GFLOP):
//   C[8192,10000] = A[8192,1024]bf16 @ B[1024,10000]bf16 + bias
// Block = 256 threads (8 waves), macro-tile 128(M) x 80(N):
//   wave w -> m-tile w, 5 n-tiles, A operand reused across 5 WMMAs.
// Double-buffered LDS slabs staged with GLOBAL_LOAD_ASYNC_TO_LDS_B128:
//   fetch of K-slab k+1 overlaps WMMA compute on slab k (ASYNCcnt-tracked).
// ~6x less L2 traffic than one-wave-per-tile -> WMMA-roofline bound.
// ---------------------------------------------------------------------------
#define DM 128
#define DN 80
__global__ void k_dense_lds(const __bf16* __restrict__ A,   // [M,K]
                            const __bf16* __restrict__ Bm,  // [K,N]
                            const float*  __restrict__ bias,
                            float*        __restrict__ C,
                            int M, int N, int K) {
  __shared__ __bf16 sA[2][DM * 32];   // row-major, stride 32 (8KB each)
  __shared__ __bf16 sB[2][32 * DN];   // row-major, stride 80 (5KB each)

  const int t    = threadIdx.x;
  const int wave = t >> 5;               // 0..7 -> m-tile within macro
  const int lane = t & 31;
  const int lo = lane & 15, hi = (lane >> 4) & 1;

  const int nblk = N / DN;               // 125
  const int R0 = (blockIdx.x / nblk) * DM;
  const int N0 = (blockIdx.x % nblk) * DN;

  // A staging: thread -> (row = t/2, half = t&1): 32B = 2 async b128
  const int arow = t >> 1, ahalf = (t & 1) * 16;
  // B staging: threads 0..159 -> (row = t/5, chunk = t%5): 32B = 2 async b128
  const int brow = t / 5, bcol = (t % 5) * 16;

  auto stage = [&](int buf, int k0) {
    {
      const __bf16* ga = A + (size_t)(R0 + arow) * K + (k0 + ahalf);
      unsigned da = (unsigned)(uintptr_t)(&sA[buf][arow * 32 + ahalf]);
      async_copy_b128(da,     ga);
      async_copy_b128(da + 16, ga + 8);     // +8 elems = +16 bytes
    }
    if (t < 160) {                          // wave-uniform (waves 0-4 vs 5-7)
      const __bf16* gb = Bm + (size_t)(k0 + brow) * N + (N0 + bcol);
      unsigned db = (unsigned)(uintptr_t)(&sB[buf][brow * DN + bcol]);
      async_copy_b128(db,      gb);
      async_copy_b128(db + 16, gb + 8);
    }
  };

  v8f acc[5];
#pragma unroll
  for (int j = 0; j < 5; ++j) acc[j] = (v8f){};

  const int ksteps = K / 32;               // 32
  stage(0, 0);
  wait_async0();
  __syncthreads();

  for (int kt = 0; kt < ksteps; ++kt) {
    const int cur = kt & 1;
    if (kt + 1 < ksteps) stage(cur ^ 1, (kt + 1) * 32);

    // ---- compute from buffer `cur`: one A operand, 5 WMMAs ----
    v16bf a;
    {
      const __bf16* p = &sA[cur][(wave * 16 + lo) * 32 + hi * 8];
      v8bf a0 = *(const v8bf*)(p);
      v8bf a1 = *(const v8bf*)(p + 16);
#pragma unroll
      for (int i = 0; i < 8; ++i) { a[i] = a0[i]; a[i + 8] = a1[i]; }
    }
#pragma unroll
    for (int j = 0; j < 5; ++j) {
      const __bf16* p = &sB[cur][(hi * 8) * DN + j * 16 + lo];
      v16bf b;
#pragma unroll
      for (int i = 0; i < 8; ++i) { b[i] = p[i * DN]; b[i + 8] = p[(16 + i) * DN]; }
      acc[j] = __builtin_amdgcn_wmma_f32_16x16x32_bf16(false, a, false, b,
                                                       (short)0, acc[j],
                                                       false, false);
    }

    // My async loads into buf^1 done; barrier also fences everyone's reads
    // of `cur` before it is overwritten in the next iteration.
    wait_async0();
    __syncthreads();
  }

  // ---- epilogue: bias + scattered f32 stores ----
#pragma unroll
  for (int j = 0; j < 5; ++j) {
    int col = N0 + j * 16 + lo;
    float bv = bias[col];
#pragma unroll
    for (int r = 0; r < 8; ++r)
      C[(size_t)(R0 + wave * 16 + r + hi * 8) * N + col] = acc[j][r] + bv;
  }
}

// ---------------------------------------------------------------------------
// One recurrent step:  g = Gx_t + h_cur @ Wh  (Gx_t already holds x@Wx+b_gate)
// s = sigmoid(g); c_nxt = c*s + tanh(g)*s; h_nxt = tanh(c_nxt)*s  (as bf16)
// ---------------------------------------------------------------------------
__global__ void k_lstm_step(const float*  __restrict__ Gx_t,   // [32,HID] f32
                            const __bf16* __restrict__ h_cur,  // [32,HID]
                            const __bf16* __restrict__ Wh,     // [HID,HID]
                            const float*  __restrict__ c_cur,  // [32,HID]
                            float*        __restrict__ c_nxt,
                            __bf16*       __restrict__ h_nxt) {
  int wave = (int)((blockIdx.x * blockDim.x + threadIdx.x) >> 5);  // 0..127
  int lane = threadIdx.x & 31;
  const int ntn = HID >> 4;                  // 64
  int mt = wave / ntn, nt = wave % ntn;
  int row0 = mt << 4, n0 = nt << 4;
  int lo = lane & 15, hi = (lane >> 4) & 1;

  v8f acc;
#pragma unroll
  for (int r = 0; r < 8; ++r)
    acc[r] = Gx_t[(size_t)(row0 + r + hi * 8) * HID + (n0 + lo)];

  for (int k0 = 0; k0 < HID; k0 += 32) {
    v16bf a = load_a_tile(h_cur, HID, row0, k0, lane);
    v16bf b = load_b_tile(Wh, HID, k0, n0, lane);
    acc = __builtin_amdgcn_wmma_f32_16x16x32_bf16(false, a, false, b,
                                                  (short)0, acc, false, false);
  }
#pragma unroll
  for (int r = 0; r < 8; ++r) {
    size_t idx = (size_t)(row0 + r + hi * 8) * HID + (n0 + lo);
    float g  = acc[r];
    float s  = 1.0f / (1.0f + __expf(-g));
    float cn = c_cur[idx] * s + tanhf(g) * s;
    c_nxt[idx] = cn;
    h_nxt[idx] = (__bf16)(tanhf(cn) * s);
  }
}

// ---------------------------------------------------------------------------
// out_t = h_nxt @ w_out + b_out  -> stored as bf16 into ys[t]
// ---------------------------------------------------------------------------
__global__ void k_out_proj(const __bf16* __restrict__ h,   // [32,HID]
                           const __bf16* __restrict__ Wo,  // [HID,HID]
                           const float*  __restrict__ bo,
                           __bf16*       __restrict__ ys_t) {
  int wave = (int)((blockIdx.x * blockDim.x + threadIdx.x) >> 5);
  int lane = threadIdx.x & 31;
  const int ntn = HID >> 4;
  int mt = wave / ntn, nt = wave % ntn;
  int row0 = mt << 4, n0 = nt << 4;
  int lo = lane & 15, hi = (lane >> 4) & 1;

  v8f acc = {};
  for (int k0 = 0; k0 < HID; k0 += 32) {
    v16bf a = load_a_tile(h, HID, row0, k0, lane);
    v16bf b = load_b_tile(Wo, HID, k0, n0, lane);
    acc = __builtin_amdgcn_wmma_f32_16x16x32_bf16(false, a, false, b,
                                                  (short)0, acc, false, false);
  }
  float bv = bo[n0 + lo];
#pragma unroll
  for (int r = 0; r < 8; ++r)
    ys_t[(size_t)(row0 + r + hi * 8) * HID + (n0 + lo)] = (__bf16)(acc[r] + bv);
}

// ---------------------------------------------------------------------------
// Support kernels
// ---------------------------------------------------------------------------
__global__ void k_f32_to_bf16(const float* __restrict__ src,
                              __bf16* __restrict__ dst, int n) {
  int i = blockIdx.x * blockDim.x + threadIdx.x;
  if (i < n) dst[i] = (__bf16)src[i];
}

// X[t*BATCH+b][e] = bf16( embedding[ seq[b][t] ][e] )   (time-major for scan)
__global__ void k_embed_gather(const int* __restrict__ seq,
                               const float* __restrict__ emb,
                               __bf16* __restrict__ X) {
  int tb = blockIdx.x;               // t*BATCH + b
  int t = tb / BATCH, b = tb % BATCH;
  int tok = seq[b * SEQ + t];
  const float* src = emb + (size_t)tok * EMB;
  __bf16* dst = X + (size_t)tb * EMB;
  for (int e = threadIdx.x; e < EMB; e += blockDim.x)
    dst[e] = (__bf16)src[e];
}

__global__ void k_zero_state(float* __restrict__ c0, __bf16* __restrict__ h0) {
  int i = blockIdx.x * blockDim.x + threadIdx.x;
  if (i < BATCH * HID) { c0[i] = 0.0f; h0[i] = (__bf16)0.0f; }
}

// ---------------------------------------------------------------------------
extern "C" void kernel_launch(void* const* d_in, const int* in_sizes, int n_in,
                              void* d_out, int out_size, void* d_ws, size_t ws_size,
                              hipStream_t stream) {
  const int*   seq    = (const int*)  d_in[0];
  const float* emb    = (const float*)d_in[1];
  const float* wgate  = (const float*)d_in[2];   // [EMB+HID, HID]
  const float* bgate  = (const float*)d_in[3];
  const float* wout   = (const float*)d_in[4];
  const float* bout   = (const float*)d_in[5];
  const float* wdense = (const float*)d_in[6];   // [HID, VOCAB]
  const float* bdense = (const float*)d_in[7];
  float* out = (float*)d_out;
  (void)in_sizes; (void)n_in; (void)out_size; (void)ws_size;

  char* p = (char*)d_ws;
  auto alloc = [&](size_t bytes) {
    char* r = p; p += (bytes + 255) & ~(size_t)255; return r;
  };
  __bf16* X  = (__bf16*)alloc((size_t)SEQ * BATCH * EMB * 2);  // 16.8 MB
  __bf16* Wx = (__bf16*)alloc((size_t)EMB * HID * 2);          //  2.1 MB
  __bf16* Wh = (__bf16*)alloc((size_t)HID * HID * 2);          //  2.1 MB
  __bf16* Wo = (__bf16*)alloc((size_t)HID * HID * 2);          //  2.1 MB
  __bf16* Wd = (__bf16*)alloc((size_t)HID * VOCAB * 2);        // 20.5 MB
  float*  Gx = (float*) alloc((size_t)SEQ * BATCH * HID * 4);  // 33.6 MB
  float*  c0 = (float*) alloc((size_t)BATCH * HID * 4);
  float*  c1 = (float*) alloc((size_t)BATCH * HID * 4);
  __bf16* h0 = (__bf16*)alloc((size_t)BATCH * HID * 2);
  __bf16* h1 = (__bf16*)alloc((size_t)BATCH * HID * 2);
  __bf16* YS = (__bf16*)alloc((size_t)SEQ * BATCH * HID * 2);  // 16.8 MB

  // One-time bf16 conversions (weights live in L2 afterwards: 192 MB covers all)
  const int nEH = EMB * HID;
  k_f32_to_bf16<<<(nEH + 255) / 256, 256, 0, stream>>>(wgate,       Wx, nEH);
  k_f32_to_bf16<<<(nEH + 255) / 256, 256, 0, stream>>>(wgate + nEH, Wh, nEH);
  k_f32_to_bf16<<<(nEH + 255) / 256, 256, 0, stream>>>(wout,        Wo, nEH);
  const int nDV = HID * VOCAB;
  k_f32_to_bf16<<<(nDV + 255) / 256, 256, 0, stream>>>(wdense, Wd, nDV);

  k_embed_gather<<<SEQ * BATCH, 256, 0, stream>>>(seq, emb, X);
  k_zero_state<<<(BATCH * HID + 255) / 256, 256, 0, stream>>>(c0, h0);

  // Hoisted input projection: Gx = X @ Wx + b_gate   [8192,1024] (17.2 GFLOP)
  {
    int tiles = (SEQ * BATCH / 16) * (HID / 16);     // 512*64 = 32768
    k_gemm_bf16_bias<<<tiles / 8, 256, 0, stream>>>(X, Wx, bgate, Gx,
                                                    SEQ * BATCH, HID, EMB);
  }

  // Sequential recurrence with ping-pong state (t=0 reads zeroed buffers 0)
  float*  cbuf[2] = { c0, c1 };
  __bf16* hbuf[2] = { h0, h1 };
  for (int t = 0; t < SEQ; ++t) {
    int cur = t & 1, nxt = cur ^ 1;
    k_lstm_step<<<32, 128, 0, stream>>>(Gx + (size_t)t * BATCH * HID,
                                        hbuf[cur], Wh, cbuf[cur],
                                        cbuf[nxt], hbuf[nxt]);
    k_out_proj<<<32, 128, 0, stream>>>(hbuf[nxt], Wo, bout,
                                       YS + (size_t)t * BATCH * HID);
  }

  // Dominant GEMM: logits = YS @ Wd + b_dense   [8192,10000] (167.8 GFLOP)
  {
    int nblocks = (SEQ * BATCH / DM) * (VOCAB / DN);   // 64*125 = 8000
    k_dense_lds<<<nblocks, 256, 0, stream>>>(YS, Wd, bdense, out,
                                             SEQ * BATCH, VOCAB, HID);
  }
}
